// MHAttention_52132313038888
// MI455X (gfx1250) — compile-verified, matrix operand order checked by reading, not run on hardware
//
#include <hip/hip_runtime.h>
#include <hip/hip_bf16.h>

// ---------------------------------------------------------------------------
// Types for CDNA5 WMMA (wave32): v_wmma_f32_16x16x32_bf16
// ---------------------------------------------------------------------------
typedef __attribute__((ext_vector_type(16))) __bf16 bf16x16;
typedef __attribute__((ext_vector_type(8)))  __bf16 bf16x8;
typedef __attribute__((ext_vector_type(8)))  float  f32x8;

union FragBF {
    bf16x16 v;
    bf16x8  h[2];
};

__device__ __forceinline__ __bf16 f2bf(float x) {
    // round-to-nearest-even fp32 -> bf16
    unsigned u = __float_as_uint(x);
    unsigned r = u + 0x7FFFu + ((u >> 16) & 1u);
    unsigned short s = (unsigned short)(r >> 16);
    __bf16 b;
    __builtin_memcpy(&b, &s, sizeof(b));
    return b;
}

__device__ __forceinline__ float bf2f(__bf16 b) {
    unsigned short s;
    __builtin_memcpy(&s, &b, sizeof(s));
    return __uint_as_float((unsigned)s << 16);
}

__device__ __forceinline__ f32x8 wmma_bf16(const FragBF& a, const FragBF& b, f32x8 c) {
    return __builtin_amdgcn_wmma_f32_16x16x32_bf16(
        /*neg_a=*/false, a.v, /*neg_b=*/false, b.v,
        /*c_mod=*/(short)0, c, /*reuse_a=*/false, /*reuse_b=*/false);
}

// ---------------------------------------------------------------------------
// CDNA5 async global->LDS copy (no VGPR round-trip; tracked by ASYNCcnt).
// LDS destination = low 32 bits of the generic shared pointer.
// Each wave must drain its own ASYNCcnt before the
// workgroup barrier so all staged tiles are visible past the barrier.
// ---------------------------------------------------------------------------
__device__ __forceinline__ void async_b128(void* lds_dst, const void* gsrc) {
    unsigned loff = (unsigned)(uintptr_t)lds_dst;
    asm volatile("global_load_async_to_lds_b128 %0, %1, off"
                 :: "v"(loff), "v"(gsrc)
                 : "memory");
}
__device__ __forceinline__ void wait_async0() {
    asm volatile("s_wait_asynccnt 0" ::: "memory");
}

// ---------------------------------------------------------------------------
// One-time conversion kernels (off the critical loop; bandwidth-trivial)
// ---------------------------------------------------------------------------
__global__ __launch_bounds__(256)
void cvt_f32_bf16(const float* __restrict__ src, __bf16* __restrict__ dst, int n8) {
    int i = blockIdx.x * 256 + threadIdx.x;
    if (i >= n8) return;
    size_t base = (size_t)i * 8;
    bf16x8 o;
#pragma unroll
    for (int j = 0; j < 8; ++j) o[j] = f2bf(src[base + j]);
    *(bf16x8*)(dst + base) = o;
}

// src [R][C] fp32 -> dst [C][R] bf16 (weights pre-transposed for B-staging)
__global__ __launch_bounds__(256)
void cvt_f32_bf16_T(const float* __restrict__ src, __bf16* __restrict__ dst,
                    int R, int C) {
    int idx = blockIdx.x * 256 + threadIdx.x;
    if (idx >= R * C) return;
    int c = idx / R, r = idx % R;              // r fastest -> dst writes coalesced
    dst[(size_t)c * R + r] = f2bf(src[(size_t)r * C + c]);
}

// bias[row][j] = -1e9 * mask[row][2j+1]  (nearest resize: n/kp==2 -> col 2j+1)
__global__ __launch_bounds__(256)
void build_bias(const float* __restrict__ mask, __bf16* __restrict__ bias) {
    int idx = blockIdx.x * 256 + threadIdx.x;   // over 4096*2048, j fastest
    int row = idx >> 11, j = idx & 2047;
    bias[idx] = f2bf(mask[(size_t)row * 4096 + 2 * j + 1] * (-1.0e9f));
}

// ---------------------------------------------------------------------------
// Tiled GEMM, bf16 in (B pre-transposed: Bt[N][K]), fp32 accumulate.
// Block tile 128x128, K-tile 64, 256 threads = 8 waves (4x2), wave tile 32x64.
// TRANSA: A element (m,k) at A[k*lda + m]  (projections).
// OUTMODE: 0 = fp32 row-major (+bias), 1 = bf16 row-major, 2 = bf16 transposed
//          (OUTMODE 2 ldc = row stride of C^T; stores are packed b128).
// All dims here are multiples of 128/128/64 -> no edge guards.
// ---------------------------------------------------------------------------
template <bool TRANSA, int OUTMODE, bool BIAS>
__global__ __launch_bounds__(256)
void gemm_bf16wmma(const __bf16* __restrict__ Ag, const __bf16* __restrict__ Btg,
                   void* __restrict__ Cg, const float* __restrict__ bias,
                   int M, int N, int K, int lda, int ldb, int ldc,
                   long long asb, long long bsb, long long csb)
{
    const __bf16* A  = Ag  + (size_t)blockIdx.z * (size_t)asb;
    const __bf16* Bt = Btg + (size_t)blockIdx.z * (size_t)bsb;

    const int M0 = blockIdx.y * 128;
    const int N0 = blockIdx.x * 128;
    const int tid   = threadIdx.x;
    const int wave  = tid >> 5;
    const int lane  = tid & 31;
    const int wm    = wave >> 1;       // 0..3  (M, 32 rows each)
    const int wn    = wave & 1;        // 0..1  (N, 64 cols each)
    const int half  = lane >> 4;
    const int col16 = lane & 15;

    __shared__ __align__(16) __bf16 As[128 * 64];   // [m][k]
    __shared__ __align__(16) __bf16 Bs[128 * 64];   // [n][k]

    f32x8 acc[2][4];
#pragma unroll
    for (int i = 0; i < 2; ++i)
#pragma unroll
        for (int t = 0; t < 4; ++t)
#pragma unroll
            for (int r = 0; r < 8; ++r) acc[i][t][r] = 0.0f;

    for (int k0 = 0; k0 < K; k0 += 64) {
        if (TRANSA) {
            // transpose staging: coalesced along m; 2-byte register copies
            for (int e = tid; e < 128 * 64; e += 256) {
                int r = e & 127, c = e >> 7;
                As[r * 64 + c] = A[(size_t)(k0 + c) * lda + (M0 + r)];
            }
        } else {
            // async b128 tile copy, no VGPR round-trip
#pragma unroll
            for (int i = 0; i < 4; ++i) {
                int ch = tid + i * 256;
                int r = ch >> 3, c8 = (ch & 7) * 8;
                async_b128(&As[r * 64 + c8], &A[(size_t)(M0 + r) * lda + k0 + c8]);
            }
        }
        // B pre-transposed: async b128 tile copy
#pragma unroll
        for (int i = 0; i < 4; ++i) {
            int ch = tid + i * 256;
            int nr = ch >> 3, c8 = (ch & 7) * 8;
            async_b128(&Bs[nr * 64 + c8], &Bt[(size_t)(N0 + nr) * ldb + k0 + c8]);
        }
        wait_async0();
        __syncthreads();

#pragma unroll
        for (int kk = 0; kk < 2; ++kk) {
            FragBF af[2], bfm[4];
#pragma unroll
            for (int i = 0; i < 2; ++i) {
                const __bf16* p = &As[(wm * 32 + i * 16 + col16) * 64 + kk * 32 + half * 8];
                af[i].h[0] = *(const bf16x8*)p;
                af[i].h[1] = *(const bf16x8*)(p + 16);
            }
#pragma unroll
            for (int t = 0; t < 4; ++t) {
                const __bf16* p = &Bs[(wn * 64 + t * 16 + col16) * 64 + kk * 32 + half * 8];
                bfm[t].h[0] = *(const bf16x8*)p;
                bfm[t].h[1] = *(const bf16x8*)(p + 16);
            }
#pragma unroll
            for (int i = 0; i < 2; ++i)
#pragma unroll
                for (int t = 0; t < 4; ++t)
                    acc[i][t] = wmma_bf16(af[i], bfm[t], acc[i][t]);
        }
        __syncthreads();
    }

    // Store. C layout: VGPR r -> row r (lanes 0-15) / r+8 (lanes 16-31)
#pragma unroll
    for (int i = 0; i < 2; ++i)
#pragma unroll
        for (int t = 0; t < 4; ++t) {
            int gcol = N0 + wn * 64 + t * 16 + col16;
            if (OUTMODE == 0) {
                float* C = (float*)Cg + (size_t)blockIdx.z * (size_t)csb;
                float bv = BIAS ? bias[gcol] : 0.0f;
#pragma unroll
                for (int r = 0; r < 8; ++r) {
                    int grow = M0 + wm * 32 + i * 16 + r + half * 8;
                    C[(size_t)grow * ldc + gcol] = acc[i][t][r] + bv;
                }
            } else if (OUTMODE == 1) {
                __bf16* C = (__bf16*)Cg + (size_t)blockIdx.z * (size_t)csb;
#pragma unroll
                for (int r = 0; r < 8; ++r) {
                    int grow = M0 + wm * 32 + i * 16 + r + half * 8;
                    C[(size_t)grow * ldc + gcol] = f2bf(acc[i][t][r]);
                }
            } else {
                // transposed bf16: rows r..r+7 are contiguous -> packed b128 store
                __bf16* C = (__bf16*)Cg + (size_t)blockIdx.z * (size_t)csb;
                bf16x8 o;
#pragma unroll
                for (int r = 0; r < 8; ++r) o[r] = f2bf(acc[i][t][r]);
                int growb = M0 + wm * 32 + i * 16 + half * 8;
                *(bf16x8*)&C[(size_t)gcol * ldc + growb] = o;
            }
        }
}

// ---------------------------------------------------------------------------
// Flash-style Linformer attention, all-bf16 operands, fp32 accumulate.
// Grid: (n/64, heads, batch), 128 threads = 4 waves; wave w owns q rows
// [w*16, w*16+16). Loops kp in 64-col chunks with online softmax.
// ---------------------------------------------------------------------------
__global__ __launch_bounds__(128)
void linformer_attn(const __bf16* __restrict__ q, const __bf16* __restrict__ kb,
                    const __bf16* __restrict__ vb, const __bf16* __restrict__ biasb,
                    __bf16* __restrict__ ctx)
{
    constexpr int N = 4096, D = 1024, KP = 2048;
    const int b     = blockIdx.z;
    const int h     = blockIdx.y;
    const int row0  = blockIdx.x * 64;
    const int tid   = threadIdx.x;
    const int wave  = tid >> 5;
    const int lane  = tid & 31;
    const int half  = lane >> 4;
    const int col16 = lane & 15;
    const int hoff  = h * 64;

    __shared__ __align__(16) __bf16 Ksh[64 * 64];   // [kp-row][dh]
    __shared__ __align__(16) __bf16 VshT[64 * 64];  // [dh][kp-row]
    __shared__ __align__(16) float  Ssh[64 * 64];   // scores tile
    __shared__ __align__(16) __bf16 Psh[64 * 64];   // exp(P) tile
    __shared__ float Mst[64], Lst[64], Fac[64];

    const __bf16* qp  = q  + (size_t)b * N  * D;
    const __bf16* kpB = kb + (size_t)b * KP * D;
    const __bf16* vpB = vb + (size_t)b * KP * D;

    // Q fragments: straight b128 loads, live in registers all kernel
    FragBF qf[2];
    {
        const __bf16* qrow = qp + (size_t)(row0 + wave * 16 + col16) * D + hoff;
#pragma unroll
        for (int kf = 0; kf < 2; ++kf) {
            qf[kf].h[0] = *(const bf16x8*)(qrow + kf * 32 + half * 8);
            qf[kf].h[1] = *(const bf16x8*)(qrow + kf * 32 + 16 + half * 8);
        }
    }

    if (tid < 64) { Mst[tid] = -3.0e38f; Lst[tid] = 0.0f; }
    f32x8 acc[4];
#pragma unroll
    for (int t = 0; t < 4; ++t)
#pragma unroll
        for (int r = 0; r < 8; ++r) acc[t][r] = 0.0f;
    __syncthreads();

    for (int c0 = 0; c0 < KP; c0 += 64) {
        // Stage K rows via async b128; V needs a transpose -> register copies
#pragma unroll
        for (int i = 0; i < 4; ++i) {
            int ch = tid + i * 128;
            int kr = ch >> 3, c8 = (ch & 7) * 8;
            size_t gaddr = (size_t)(c0 + kr) * D + hoff + c8;
            async_b128(&Ksh[kr * 64 + c8], &kpB[gaddr]);
            bf16x8 vv = *(const bf16x8*)&vpB[gaddr];
#pragma unroll
            for (int j = 0; j < 8; ++j) VshT[(c8 + j) * 64 + kr] = vv[j];
        }
        wait_async0();
        __syncthreads();

        // S = Q @ K^T  (K dim = dh = 64 -> 2 wmma per 16x16 tile)
#pragma unroll
        for (int t = 0; t < 4; ++t) {
            f32x8 s;
#pragma unroll
            for (int r = 0; r < 8; ++r) s[r] = 0.0f;
#pragma unroll
            for (int kf = 0; kf < 2; ++kf) {
                FragBF bk;
                const __bf16* p = &Ksh[(t * 16 + col16) * 64 + kf * 32 + half * 8];
                bk.h[0] = *(const bf16x8*)p;
                bk.h[1] = *(const bf16x8*)(p + 16);
                s = wmma_bf16(qf[kf], bk, s);
            }
#pragma unroll
            for (int r = 0; r < 8; ++r) {
                int rl   = wave * 16 + r + half * 8;
                int grow = row0 + rl;
                int gcol = c0 + t * 16 + col16;
                float sv = s[r] * 0.125f + bf2f(biasb[(size_t)grow * KP + gcol]);
                Ssh[rl * 64 + t * 16 + col16] = sv;
            }
        }
        __syncthreads();

        // Online softmax update: one thread per q row
        if (tid < 64) {
            int rr = tid;
            float mold = Mst[rr];
            float mx = mold;
            for (int c = 0; c < 64; ++c) mx = fmaxf(mx, Ssh[rr * 64 + c]);
            float fac = __expf(mold - mx);
            float sum = 0.0f;
            for (int c = 0; c < 64; ++c) {
                float e = __expf(Ssh[rr * 64 + c] - mx);
                sum += e;
                Psh[rr * 64 + c] = f2bf(e);
            }
            Lst[rr] = Lst[rr] * fac + sum;
            Mst[rr] = mx;
            Fac[rr] = fac;
        }
        __syncthreads();

        // Rescale accumulators, then O += P @ V
#pragma unroll
        for (int t = 0; t < 4; ++t)
#pragma unroll
            for (int r = 0; r < 8; ++r)
                acc[t][r] *= Fac[wave * 16 + r + half * 8];

#pragma unroll
        for (int kf = 0; kf < 2; ++kf) {
            FragBF pa;
            const __bf16* pp = &Psh[(wave * 16 + col16) * 64 + kf * 32 + half * 8];
            pa.h[0] = *(const bf16x8*)pp;
            pa.h[1] = *(const bf16x8*)(pp + 16);
#pragma unroll
            for (int t = 0; t < 4; ++t) {
                FragBF bv;
                const __bf16* vp2 = &VshT[(t * 16 + col16) * 64 + kf * 32 + half * 8];
                bv.h[0] = *(const bf16x8*)vp2;
                bv.h[1] = *(const bf16x8*)(vp2 + 16);
                acc[t] = wmma_bf16(pa, bv, acc[t]);
            }
        }
        __syncthreads();
    }

    // Normalize, write ctx bf16 in [b, n, h*dh + dc] layout
#pragma unroll
    for (int t = 0; t < 4; ++t)
#pragma unroll
        for (int r = 0; r < 8; ++r) {
            int rl = wave * 16 + r + half * 8;
            float o = acc[t][r] / Lst[rl];
            ctx[((size_t)b * N + row0 + rl) * D + hoff + t * 16 + col16] = f2bf(o);
        }
}

// ---------------------------------------------------------------------------
// Host orchestration
// ---------------------------------------------------------------------------
extern "C" void kernel_launch(void* const* d_in, const int* in_sizes, int n_in,
                              void* d_out, int out_size, void* d_ws, size_t ws_size,
                              hipStream_t stream)
{
    (void)in_sizes; (void)n_in; (void)out_size; (void)ws_size;
    constexpr int B = 2, N = 4096, D = 1024, KP = 2048;
    constexpr size_t ND  = (size_t)B * N * D;    // 8388608
    constexpr size_t WSZ = (size_t)D * D;        // 1048576
    constexpr size_t KD  = (size_t)B * KP * D;   // 4194304

    const float* x    = (const float*)d_in[0];
    const float* mask = (const float*)d_in[1];
    const float* Wq   = (const float*)d_in[2];
    const float* Wk   = (const float*)d_in[3];
    const float* Wv   = (const float*)d_in[4];
    const float* Wo   = (const float*)d_in[5];
    const float* bo   = (const float*)d_in[6];
    const float* pk   = (const float*)d_in[7];
    const float* pv   = (const float*)d_in[8];
    float* out = (float*)d_out;

    __bf16* wsb  = (__bf16*)d_ws;
    __bf16* xb   = wsb;             // ND
    __bf16* wqb  = xb   + ND;       // WSZ  (transposed [N][K])
    __bf16* wkb  = wqb  + WSZ;
    __bf16* wvb  = wkb  + WSZ;
    __bf16* wob  = wvb  + WSZ;
    __bf16* pkb  = wob  + WSZ;      // ND   (reused as bias after step 2)
    __bf16* pvb  = pkb  + ND;       // ND
    __bf16* tmpt = pvb  + ND;       // ND   (xW transposed [1024][8192]; reused as ctx)
    __bf16* kbuf = tmpt + ND;       // KD
    __bf16* vbuf = kbuf + KD;       // KD
    __bf16* qb   = vbuf + KD;       // ND
    __bf16* bias = pkb;             // reuse
    __bf16* ctx  = tmpt;            // reuse

    dim3 b256(256);
    // --- one-time conversions ---
    cvt_f32_bf16<<<dim3(ND / 8 / 256), b256, 0, stream>>>(x,  xb,  (int)(ND / 8));
    cvt_f32_bf16_T<<<dim3(WSZ / 256), b256, 0, stream>>>(Wq, wqb, D, D);
    cvt_f32_bf16_T<<<dim3(WSZ / 256), b256, 0, stream>>>(Wk, wkb, D, D);
    cvt_f32_bf16_T<<<dim3(WSZ / 256), b256, 0, stream>>>(Wv, wvb, D, D);
    cvt_f32_bf16_T<<<dim3(WSZ / 256), b256, 0, stream>>>(Wo, wob, D, D);
    cvt_f32_bf16<<<dim3(ND / 8 / 256), b256, 0, stream>>>(pk, pkb, (int)(ND / 8));
    cvt_f32_bf16<<<dim3(ND / 8 / 256), b256, 0, stream>>>(pv, pvb, (int)(ND / 8));

    dim3 gXW(D / 128, (B * N) / 128, 1);         // (8, 64)
    dim3 gPR(D / 128, KP / 128, B);              // (8, 16, 2)

    // 1. tmpt = T(xb @ Wk)      -> bf16 [1024][8192]
    gemm_bf16wmma<false, 2, false><<<gXW, b256, 0, stream>>>(
        xb, wkb, tmpt, nullptr, B * N, D, D, D, D, B * N, 0, 0, 0);
    // 2. kbuf[b] = proj_k^T @ tmp[b]   (B already transposed in tmpt)
    gemm_bf16wmma<true, 1, false><<<gPR, b256, 0, stream>>>(
        pkb, tmpt, kbuf, nullptr, KP, D, N, KP, B * N, D,
        0, (long long)N, (long long)KP * D);
    // 3. tmpt = T(xb @ Wv)
    gemm_bf16wmma<false, 2, false><<<gXW, b256, 0, stream>>>(
        xb, wvb, tmpt, nullptr, B * N, D, D, D, D, B * N, 0, 0, 0);
    // 4. vbuf[b] = proj_v^T @ tmp[b]
    gemm_bf16wmma<true, 1, false><<<gPR, b256, 0, stream>>>(
        pvb, tmpt, vbuf, nullptr, KP, D, N, KP, B * N, D,
        0, (long long)N, (long long)KP * D);
    // 5. qb = xb @ Wq           -> bf16 row-major
    gemm_bf16wmma<false, 1, false><<<gXW, b256, 0, stream>>>(
        xb, wqb, qb, nullptr, B * N, D, D, D, D, D, 0, 0, 0);
    // 6. bias (reuses pkb region; pkb last read in step 2)
    build_bias<<<dim3((N * KP) / 256), b256, 0, stream>>>(mask, bias);
    // 7. attention -> ctx (reuses tmpt; last read in step 4)
    dim3 gAT(N / 64, 16, B);
    linformer_attn<<<gAT, dim3(128), 0, stream>>>(qb, kbuf, vbuf, bias, ctx);
    // 8. out = ctx @ Wo + bo    -> fp32
    gemm_bf16wmma<false, 0, true><<<gXW, b256, 0, stream>>>(
        ctx, wob, out, bo, B * N, D, D, D, D, D, 0, 0, 0);
}
